// GraphSAGE_13786845020247
// MI455X (gfx1250) — matrix operand order repaired and checked by reading, not compiled, not run on hardware
//
#include <hip/hip_runtime.h>
#include <hip/hip_bf16.h>

typedef __attribute__((ext_vector_type(16))) __bf16 v16bf;
typedef __attribute__((ext_vector_type(8)))  __bf16 v8bf;
typedef __attribute__((ext_vector_type(8)))  float  v8f;

#define DIM   128
#define KNBR  16
#define NPB   128   // nodes per block (8 waves x 16)

// LDS map (dynamic shared, base offset 0: no static LDS in this kernel)
#define LDS_W_OFF   0          // [Wih|Whh] 1024 rows x 128 bf16 = 262144 B
#define LDS_XT_OFF  262144     // 128 x 128 bf16 = 32768 B
#define LDS_H_OFF   294912     // 128 x 128 bf16 = 32768 B

// Async DMA: global (SGPR base + 32-bit lane offset) -> LDS, 16B per lane.
// IOFFSET is applied to BOTH the LDS and the global address (ISA 08 §4.4).
#define ASYNC_B128(ldsoff, goff, sbase, IMM)                                   \
    asm volatile("global_load_async_to_lds_b128 %0, %1, %2 offset:" #IMM      \
                 :: "v"(ldsoff), "v"(goff), "s"(sbase) : "memory")

#define WAIT_ASYNC() asm volatile("s_wait_asynccnt 0x0" ::: "memory")

__device__ __forceinline__ float sigm_f(float x) { return 1.0f / (1.0f + __expf(-x)); }
__device__ __forceinline__ float tanh_f(float x) {
    float e = __expf(-2.0f * x);
    return (1.0f - e) / (1.0f + e);
}

// ---------------------------------------------------------------------------
// Prep: f32 -> bf16 conversions + bias fold.  wg = [Wih(512x128)|Whh(512x128)],
// wo = [Wself(128x128)|Wneigh(128x128)], all row-major [n][k] (B-fragment order).
// ---------------------------------------------------------------------------
__global__ void sage_prep_kernel(const float* __restrict__ in_feat,
                                 const float* __restrict__ Wih1, const float* __restrict__ Whh1,
                                 const float* __restrict__ bih1, const float* __restrict__ bhh1,
                                 const float* __restrict__ Wself1, const float* __restrict__ Wneigh1,
                                 const float* __restrict__ Wih2, const float* __restrict__ Whh2,
                                 const float* __restrict__ bih2, const float* __restrict__ bhh2,
                                 const float* __restrict__ Wself2, const float* __restrict__ Wneigh2,
                                 __bf16* __restrict__ x_bf,
                                 __bf16* __restrict__ wg1, __bf16* __restrict__ wg2,
                                 __bf16* __restrict__ wo1, __bf16* __restrict__ wo2,
                                 float* __restrict__ bc1, float* __restrict__ bc2,
                                 long total_x) {
    long i = (long)blockIdx.x * blockDim.x + threadIdx.x;
    if (i < total_x) x_bf[i] = (__bf16)in_feat[i];
    if (i < 4 * DIM * DIM) {              // 65536: Wih/Whh
        wg1[i] = (__bf16)Wih1[i];  wg1[4 * DIM * DIM + i] = (__bf16)Whh1[i];
        wg2[i] = (__bf16)Wih2[i];  wg2[4 * DIM * DIM + i] = (__bf16)Whh2[i];
    }
    if (i < DIM * DIM) {                  // 16384: Wself/Wneigh
        wo1[i] = (__bf16)Wself1[i];  wo1[DIM * DIM + i] = (__bf16)Wneigh1[i];
        wo2[i] = (__bf16)Wself2[i];  wo2[DIM * DIM + i] = (__bf16)Wneigh2[i];
    }
    if (i < 4 * DIM) { bc1[i] = bih1[i] + bhh1[i]; bc2[i] = bih2[i] + bhh2[i]; }
}

// ---------------------------------------------------------------------------
// SAGE layer: LSTM aggregation over K=16 gathered neighbors + output matmul.
// 320KB dynamic LDS: [Wih|Whh bf16 256KB][xt 32KB][h 32KB].
// ---------------------------------------------------------------------------
__global__ __launch_bounds__(256, 1) void sage_layer_kernel(
        const __bf16* __restrict__ x_bf,   // [N,128] bf16
        const int*    __restrict__ nbr,    // [N,16]
        const __bf16* __restrict__ wg,     // [Wih|Whh] rows 0..1023, 128 cols
        const float*  __restrict__ bcomb,  // [512] bih+bhh
        const __bf16* __restrict__ wo,     // [Wself|Wneigh] rows 0..255
        const float*  __restrict__ bself,  // [128]
        float*        __restrict__ y,      // [N,128] pre-BN output
        int N) {
    extern __shared__ char smem[];
    __bf16* lds_w  = (__bf16*)(smem + LDS_W_OFF);
    __bf16* lds_xt = (__bf16*)(smem + LDS_XT_OFF);
    __bf16* lds_h  = (__bf16*)(smem + LDS_H_OFF);

    const int tid  = threadIdx.x;
    const int lane = tid & 31;
    const int wave = tid >> 5;       // 0..7
    const int half = lane >> 4;      // 0/1
    const int l    = lane & 15;
    const int base = blockIdx.x * NPB;
    const int grow = tid >> 1;       // gather row 0..127
    const int gsub = tid & 1;        // half-row (64 bf16 = 128 B each)

    // ---- async DMA: weights (256 KB) straight into LDS, zero VGPR payload
    {
        unsigned off0 = (unsigned)(tid * 16);
#pragma unroll 1
        for (int j = 0; j < 64; ++j) {
            ASYNC_B128(off0, off0, wg, 0);
            off0 += 4096;
        }
    }
    // ---- zero h staging
    {
        uint4 z; z.x = z.y = z.z = z.w = 0u;
        uint4* dh = (uint4*)lds_h;
#pragma unroll
        for (int j = 0; j < 8; ++j) dh[tid + j * 256] = z;
    }
    // ---- async gather for timestep 0
    {
        int node = base + grow; if (node >= N) node = N - 1;
        int nb = nbr[node * KNBR + 0];
        unsigned go = (unsigned)(nb * (DIM * 2) + gsub * 128);
        unsigned lo = (unsigned)(LDS_XT_OFF + grow * (DIM * 2) + gsub * 128);
        ASYNC_B128(lo, go, x_bf, 0);   ASYNC_B128(lo, go, x_bf, 16);
        ASYNC_B128(lo, go, x_bf, 32);  ASYNC_B128(lo, go, x_bf, 48);
        ASYNC_B128(lo, go, x_bf, 64);  ASYNC_B128(lo, go, x_bf, 80);
        ASYNC_B128(lo, go, x_bf, 96);  ASYNC_B128(lo, go, x_bf, 112);
    }
    WAIT_ASYNC();
    __syncthreads();

    // per-wave persistent state: c for 16 nodes x 128 cols (8 tiles, f32)
    v8f cst[8];
#pragma unroll
    for (int d = 0; d < 8; ++d)
#pragma unroll
        for (int e = 0; e < 8; ++e) cst[d][e] = 0.0f;

    // bias per (gate, col-block) for this lane's column (scalars, used post-WMMA)
    float bgd[4][8];
#pragma unroll
    for (int g = 0; g < 4; ++g)
#pragma unroll
        for (int d = 0; d < 8; ++d) bgd[g][d] = bcomb[g * DIM + d * 16 + l];

    const int wrow = wave * 16 + l;                       // A row (node in block)
    const __bf16* xbase = lds_xt + wrow * DIM + half * 16; // A frags: +kc*32 (+8)
    const __bf16* hbase = lds_h  + wrow * DIM + half * 16;
    __bf16* hstore = lds_h + (wave * 16 + 8 * half) * DIM + l; // C tile row base
    const __bf16* wlane = lds_w + l * DIM + half * 16;    // B frags: +(n-l)*DIM

    v8f z8;
#pragma unroll
    for (int e = 0; e < 8; ++e) z8[e] = 0.0f;

#pragma unroll 1
    for (int t = 0; t < KNBR; ++t) {
        // ---- load A fragments (contiguous K = half*16 + s, matches B)
        v16bf xa[4], ha[4];
#pragma unroll
        for (int kc = 0; kc < 4; ++kc) {
            v8bf xlo = *(const v8bf*)(xbase + kc * 32);
            v8bf xhi = *(const v8bf*)(xbase + kc * 32 + 8);
            v8bf hlo = *(const v8bf*)(hbase + kc * 32);
            v8bf hhi = *(const v8bf*)(hbase + kc * 32 + 8);
            xa[kc] = __builtin_shufflevector(xlo, xhi, 0,1,2,3,4,5,6,7,8,9,10,11,12,13,14,15);
            ha[kc] = __builtin_shufflevector(hlo, hhi, 0,1,2,3,4,5,6,7,8,9,10,11,12,13,14,15);
        }
        __syncthreads();   // all waves consumed xt; safe to refill

        // ---- async gather of next timestep overlaps the WMMA block below
        const bool have_next = (t + 1 < KNBR);
        if (have_next) {
            int node = base + grow; if (node >= N) node = N - 1;
            int nb = nbr[node * KNBR + (t + 1)];
            unsigned go = (unsigned)(nb * (DIM * 2) + gsub * 128);
            unsigned lo = (unsigned)(LDS_XT_OFF + grow * (DIM * 2) + gsub * 128);
            ASYNC_B128(lo, go, x_bf, 0);   ASYNC_B128(lo, go, x_bf, 16);
            ASYNC_B128(lo, go, x_bf, 32);  ASYNC_B128(lo, go, x_bf, 48);
            ASYNC_B128(lo, go, x_bf, 64);  ASYNC_B128(lo, go, x_bf, 80);
            ASYNC_B128(lo, go, x_bf, 96);  ASYNC_B128(lo, go, x_bf, 112);
        }

        // ---- gates: z = xt*Wih^T + h*Whh^T (+bias later).  i,f,g first.
#pragma unroll
        for (int d = 0; d < 8; ++d) {
            v8f acc[3];
#pragma unroll
            for (int gg = 0; gg < 3; ++gg) {
                const __bf16* wb = wlane + (gg * DIM + d * 16) * DIM;
                v8f a0 = z8, a1 = z8;   // two independent chains for ILP
#pragma unroll
                for (int kc = 0; kc < 2; ++kc) {
                    v16bf b0 = *(const v16bf*)(wb + kc * 32);
                    a0 = __builtin_amdgcn_wmma_f32_16x16x32_bf16(false, xa[kc], false, b0,
                                                                 (short)0, a0, false, false);
                    v16bf b1 = *(const v16bf*)(wb + (kc + 2) * 32);
                    a1 = __builtin_amdgcn_wmma_f32_16x16x32_bf16(false, xa[kc + 2], false, b1,
                                                                 (short)0, a1, false, false);
                }
#pragma unroll
                for (int kc = 0; kc < 2; ++kc) {
                    v16bf b0 = *(const v16bf*)(wb + 512 * DIM + kc * 32);
                    a0 = __builtin_amdgcn_wmma_f32_16x16x32_bf16(false, ha[kc], false, b0,
                                                                 (short)0, a0, false, false);
                    v16bf b1 = *(const v16bf*)(wb + 512 * DIM + (kc + 2) * 32);
                    a1 = __builtin_amdgcn_wmma_f32_16x16x32_bf16(false, ha[kc + 2], false, b1,
                                                                 (short)0, a1, false, false);
                }
                acc[gg] = a0 + a1;
            }
            // ---- c update (C layout: row = e + 8*half, col = l)
            {
                const float bi = bgd[0][d], bf_ = bgd[1][d], bg_ = bgd[2][d];
#pragma unroll
                for (int e = 0; e < 8; ++e) {
                    float iv = sigm_f(acc[0][e] + bi);
                    float fv = sigm_f(acc[1][e] + bf_);
                    float gv = tanh_f(acc[2][e] + bg_);
                    cst[d][e] = fv * cst[d][e] + iv * gv;
                }
            }
            // ---- o gate, then h = o * tanh(c), stored straight to LDS (bf16)
            {
                const __bf16* wb = wlane + (3 * DIM + d * 16) * DIM;
                v8f a0 = z8, a1 = z8;
#pragma unroll
                for (int kc = 0; kc < 2; ++kc) {
                    v16bf b0 = *(const v16bf*)(wb + kc * 32);
                    a0 = __builtin_amdgcn_wmma_f32_16x16x32_bf16(false, xa[kc], false, b0,
                                                                 (short)0, a0, false, false);
                    v16bf b1 = *(const v16bf*)(wb + (kc + 2) * 32);
                    a1 = __builtin_amdgcn_wmma_f32_16x16x32_bf16(false, xa[kc + 2], false, b1,
                                                                 (short)0, a1, false, false);
                }
#pragma unroll
                for (int kc = 0; kc < 2; ++kc) {
                    v16bf b0 = *(const v16bf*)(wb + 512 * DIM + kc * 32);
                    a0 = __builtin_amdgcn_wmma_f32_16x16x32_bf16(false, ha[kc], false, b0,
                                                                 (short)0, a0, false, false);
                    v16bf b1 = *(const v16bf*)(wb + 512 * DIM + (kc + 2) * 32);
                    a1 = __builtin_amdgcn_wmma_f32_16x16x32_bf16(false, ha[kc + 2], false, b1,
                                                                 (short)0, a1, false, false);
                }
                v8f ao = a0 + a1;
                const float bo = bgd[3][d];
#pragma unroll
                for (int e = 0; e < 8; ++e) {
                    float hv = sigm_f(ao[e] + bo) * tanh_f(cst[d][e]);
                    hstore[e * DIM + d * 16] = (__bf16)hv;
                }
            }
        }

        if (have_next) WAIT_ASYNC();   // gather landed before next-step reads
        __syncthreads();
    }

    // ---- stage self features into xt staging (async again)
    {
        int node = base + grow; if (node >= N) node = N - 1;
        unsigned go = (unsigned)(node * (DIM * 2) + gsub * 128);
        unsigned lo = (unsigned)(LDS_XT_OFF + grow * (DIM * 2) + gsub * 128);
        ASYNC_B128(lo, go, x_bf, 0);   ASYNC_B128(lo, go, x_bf, 16);
        ASYNC_B128(lo, go, x_bf, 32);  ASYNC_B128(lo, go, x_bf, 48);
        ASYNC_B128(lo, go, x_bf, 64);  ASYNC_B128(lo, go, x_bf, 80);
        ASYNC_B128(lo, go, x_bf, 96);  ASYNC_B128(lo, go, x_bf, 112);
    }
    WAIT_ASYNC();
    __syncthreads();

    // ---- output: y = x*Wself^T + bself + h*Wneigh^T  (B frags from L2)
    v16bf xs[4], hf[4];
#pragma unroll
    for (int kc = 0; kc < 4; ++kc) {
        v8bf xlo = *(const v8bf*)(xbase + kc * 32);
        v8bf xhi = *(const v8bf*)(xbase + kc * 32 + 8);
        v8bf hlo = *(const v8bf*)(hbase + kc * 32);
        v8bf hhi = *(const v8bf*)(hbase + kc * 32 + 8);
        xs[kc] = __builtin_shufflevector(xlo, xhi, 0,1,2,3,4,5,6,7,8,9,10,11,12,13,14,15);
        hf[kc] = __builtin_shufflevector(hlo, hhi, 0,1,2,3,4,5,6,7,8,9,10,11,12,13,14,15);
    }
#pragma unroll
    for (int d = 0; d < 8; ++d) {
        const int n = d * 16 + l;
        const __bf16* wb = wo + n * DIM + half * 16;
        v8f a0 = z8, a1 = z8;
#pragma unroll
        for (int kc = 0; kc < 2; ++kc) {
            v16bf b0 = *(const v16bf*)(wb + kc * 32);
            a0 = __builtin_amdgcn_wmma_f32_16x16x32_bf16(false, xs[kc], false, b0,
                                                         (short)0, a0, false, false);
            v16bf b1 = *(const v16bf*)(wb + (kc + 2) * 32);
            a1 = __builtin_amdgcn_wmma_f32_16x16x32_bf16(false, xs[kc + 2], false, b1,
                                                         (short)0, a1, false, false);
        }
#pragma unroll
        for (int kc = 0; kc < 2; ++kc) {
            v16bf b0 = *(const v16bf*)(wb + DIM * DIM + kc * 32);
            a0 = __builtin_amdgcn_wmma_f32_16x16x32_bf16(false, hf[kc], false, b0,
                                                         (short)0, a0, false, false);
            v16bf b1 = *(const v16bf*)(wb + DIM * DIM + (kc + 2) * 32);
            a1 = __builtin_amdgcn_wmma_f32_16x16x32_bf16(false, hf[kc + 2], false, b1,
                                                         (short)0, a1, false, false);
        }
        v8f a = a0 + a1;
        const float bv = bself[n];
#pragma unroll
        for (int e = 0; e < 8; ++e) {
            int node = base + wave * 16 + e + 8 * half;
            if (node < N) y[(long)node * DIM + d * 16 + l] = a[e] + bv;
        }
    }
}

// ---------------------------------------------------------------------------
// Deterministic per-column BN statistics (128 blocks, one column each).
// ---------------------------------------------------------------------------
__global__ void sage_stats_kernel(const float* __restrict__ y,
                                  float* __restrict__ ssum, float* __restrict__ ssq, int N) {
    __shared__ float s0[256], s1[256];
    const int col = blockIdx.x, tid = threadIdx.x;
    float a = 0.0f, b = 0.0f;
    for (int i = tid; i < N; i += 256) {
        float v = y[(long)i * DIM + col];
        a += v; b += v * v;
    }
    s0[tid] = a; s1[tid] = b;
    __syncthreads();
    for (int off = 128; off > 0; off >>= 1) {
        if (tid < off) { s0[tid] += s0[tid + off]; s1[tid] += s1[tid + off]; }
        __syncthreads();
    }
    if (tid == 0) { ssum[col] = s0[0]; ssq[col] = s1[0]; }
}

__global__ void sage_bn_relu_bf16_kernel(const float* __restrict__ y,
                                         const float* __restrict__ ssum, const float* __restrict__ ssq,
                                         const float* __restrict__ gamma, const float* __restrict__ beta,
                                         __bf16* __restrict__ out, int N) {
    long i = (long)blockIdx.x * blockDim.x + threadIdx.x;
    if (i >= (long)N * DIM) return;
    int col = (int)(i & (DIM - 1));
    float inv_n = 1.0f / (float)N;
    float mu = ssum[col] * inv_n;
    float var = ssq[col] * inv_n - mu * mu;
    float v = (y[i] - mu) * rsqrtf(var + 1e-5f) * gamma[col] + beta[col];
    out[i] = (__bf16)fmaxf(v, 0.0f);
}

__global__ void sage_bn_f32_kernel(const float* __restrict__ y,
                                   const float* __restrict__ ssum, const float* __restrict__ ssq,
                                   const float* __restrict__ gamma, const float* __restrict__ beta,
                                   float* __restrict__ out, int N) {
    long i = (long)blockIdx.x * blockDim.x + threadIdx.x;
    if (i >= (long)N * DIM) return;
    int col = (int)(i & (DIM - 1));
    float inv_n = 1.0f / (float)N;
    float mu = ssum[col] * inv_n;
    float var = ssq[col] * inv_n - mu * mu;
    out[i] = (y[i] - mu) * rsqrtf(var + 1e-5f) * gamma[col] + beta[col];
}

// ---------------------------------------------------------------------------
extern "C" void kernel_launch(void* const* d_in, const int* in_sizes, int n_in,
                              void* d_out, int out_size, void* d_ws, size_t ws_size,
                              hipStream_t stream) {
    const float* in_feat = (const float*)d_in[0];
    const int*   nbr     = (const int*)d_in[1];
    const float* Wih1 = (const float*)d_in[2];  const float* Whh1 = (const float*)d_in[3];
    const float* bih1 = (const float*)d_in[4];  const float* bhh1 = (const float*)d_in[5];
    const float* Wself1 = (const float*)d_in[6]; const float* bself1 = (const float*)d_in[7];
    const float* Wneigh1 = (const float*)d_in[8];
    const float* gamma1 = (const float*)d_in[9]; const float* beta1 = (const float*)d_in[10];
    const float* Wih2 = (const float*)d_in[11]; const float* Whh2 = (const float*)d_in[12];
    const float* bih2 = (const float*)d_in[13]; const float* bhh2 = (const float*)d_in[14];
    const float* Wself2 = (const float*)d_in[15]; const float* bself2 = (const float*)d_in[16];
    const float* Wneigh2 = (const float*)d_in[17];
    const float* gamma2 = (const float*)d_in[18]; const float* beta2 = (const float*)d_in[19];

    const int N = in_sizes[0] / DIM;

    // workspace carve-up (256B aligned)
    char* ws = (char*)d_ws;
    size_t off = 0;
    auto take = [&](size_t bytes) -> char* {
        char* p = ws + off;
        off += (bytes + 255) & ~(size_t)255;
        return p;
    };
    __bf16* x_bf = (__bf16*)take((size_t)N * DIM * 2);
    float*  ybuf = (float*)take((size_t)N * DIM * 4);
    __bf16* wg1  = (__bf16*)take((size_t)8 * DIM * DIM * 2);
    __bf16* wg2  = (__bf16*)take((size_t)8 * DIM * DIM * 2);
    __bf16* wo1  = (__bf16*)take((size_t)2 * DIM * DIM * 2);
    __bf16* wo2  = (__bf16*)take((size_t)2 * DIM * DIM * 2);
    float*  bc1  = (float*)take(4 * DIM * 4);
    float*  bc2  = (float*)take(4 * DIM * 4);
    float*  s1   = (float*)take(DIM * 4);
    float*  q1   = (float*)take(DIM * 4);
    float*  s2   = (float*)take(DIM * 4);
    float*  q2   = (float*)take(DIM * 4);

    const long total_x = (long)N * DIM;
    const int elem_blocks = (int)((total_x + 255) / 256);
    const int node_blocks = (N + NPB - 1) / NPB;
    const size_t lds_bytes = 320 * 1024;

    sage_prep_kernel<<<elem_blocks, 256, 0, stream>>>(
        in_feat, Wih1, Whh1, bih1, bhh1, Wself1, Wneigh1,
        Wih2, Whh2, bih2, bhh2, Wself2, Wneigh2,
        x_bf, wg1, wg2, wo1, wo2, bc1, bc2, total_x);

    // ---- layer 1
    sage_layer_kernel<<<node_blocks, 256, lds_bytes, stream>>>(
        x_bf, nbr, wg1, bc1, wo1, bself1, ybuf, N);
    sage_stats_kernel<<<DIM, 256, 0, stream>>>(ybuf, s1, q1, N);
    sage_bn_relu_bf16_kernel<<<elem_blocks, 256, 0, stream>>>(
        ybuf, s1, q1, gamma1, beta1, x_bf, N);   // x_bf now holds layer-2 input

    // ---- layer 2
    sage_layer_kernel<<<node_blocks, 256, lds_bytes, stream>>>(
        x_bf, nbr, wg2, bc2, wo2, bself2, ybuf, N);
    sage_stats_kernel<<<DIM, 256, 0, stream>>>(ybuf, s2, q2, N);
    sage_bn_f32_kernel<<<elem_blocks, 256, 0, stream>>>(
        ybuf, s2, q2, gamma2, beta2, (float*)d_out, N);
}
